// SwinBlock_60533269069926
// MI455X (gfx1250) — matrix operand order repaired and checked by reading, not compiled, not run on hardware
//
#include <hip/hip_runtime.h>

typedef __bf16 bf16;
typedef __attribute__((ext_vector_type(16))) __bf16 v16bf;
typedef __attribute__((ext_vector_type(8)))  float  v8f;
typedef __attribute__((ext_vector_type(4)))  int    v4i;

#define DIM    192
#define HIDDEN 768
#define NH     6
#define HD     32
#define HH     56
#define WW_    56
#define WS_    7
#define SS_    3
#define NWIN   2048      // 32 images * 8*8 windows
#define NTOK   49
#define NPAD   64
#define BIMG   32

#if defined(__HIP_DEVICE_COMPILE__) && __has_builtin(__builtin_amdgcn_global_load_async_to_lds_b128)
#define HAVE_ASYNC_LDS 1
#endif

// ---------------- WMMA fragment helpers (bf16, 16x16x32) ----------------
// A: 16x32 row-major source. Lane l<16 -> row l, elems 0..7 = K+8*hi.., elems 8..15 = K+16+8*hi..
__device__ __forceinline__ v16bf load_a(const bf16* base, int ld, int row0, int k0, int lane) {
    int r  = row0 + (lane & 15);
    int hi = (lane >> 4) * 8;
    const bf16* p = base + r * ld + k0 + hi;
    v16bf a;
#pragma unroll
    for (int j = 0; j < 8; ++j) a[j] = p[j];
#pragma unroll
    for (int j = 0; j < 8; ++j) a[8 + j] = p[16 + j];
    return a;
}

// B: 32x16, source stored column-major as wt[N][K] (i.e. [out][in]). Lane l<16 -> col n0+l;
// elems 0..7 = K+8*hi.., elems 8..15 = K+16+8*hi.. (same k order as load_a).
__device__ __forceinline__ v16bf load_b(const bf16* wt, int ldk, int n0, int k0, int lane) {
    int n  = n0 + (lane & 15);
    int kk = k0 + (lane >> 4) * 8;
    const bf16* p = wt + n * ldk + kk;
    v16bf b;
#pragma unroll
    for (int j = 0; j < 8; ++j) b[j] = p[j];
#pragma unroll
    for (int j = 0; j < 8; ++j) b[8 + j] = p[16 + j];
    return b;
}

__device__ __forceinline__ v8f wmma_bf16(v16bf a, v16bf b, v8f c) {
    return __builtin_amdgcn_wmma_f32_16x16x32_bf16(false, a, false, b, (short)0, c, false, false);
}

// 16-byte global -> LDS copy, async (CDNA5 ASYNCcnt path) when available
__device__ __forceinline__ void copy16_g2l(const bf16* g, bf16* l) {
#ifdef HAVE_ASYNC_LDS
    typedef __attribute__((address_space(1))) v4i* gp_t;   // ptr to AS1 v4i
    typedef __attribute__((address_space(3))) v4i* lp_t;   // ptr to AS3 v4i
    __builtin_amdgcn_global_load_async_to_lds_b128((gp_t)g, (lp_t)l, 0, 0);
#else
    *reinterpret_cast<uint4*>(l) = *reinterpret_cast<const uint4*>(g);
#endif
}

__device__ __forceinline__ void stage_fence() {
#ifdef HAVE_ASYNC_LDS
#if __has_builtin(__builtin_amdgcn_s_wait_asynccnt)
    __builtin_amdgcn_s_wait_asynccnt(0);
#endif
#endif
    __syncthreads();
}

// ---------------- weight prep: transpose to [out][in] + bf16 ----------------
__global__ void prep_weights_kernel(const float* __restrict__ qkv_w, const float* __restrict__ proj_w,
                                    const float* __restrict__ w1,    const float* __restrict__ w2,
                                    bf16* qkv_wt, bf16* proj_wt, bf16* w1t, bf16* w2t) {
    int i = blockIdx.x * 256 + threadIdx.x;
    const int N1 = 576 * 192, N2 = 192 * 192, N3 = 768 * 192;
    if (i < N1) {
        int o = i / 192, c = i % 192;
        qkv_wt[i] = (bf16)qkv_w[c * 576 + o];
    } else if (i < N1 + N2) {
        int j = i - N1; int o = j / 192, c = j % 192;
        proj_wt[j] = (bf16)proj_w[c * 192 + o];
    } else if (i < N1 + N2 + N3) {
        int j = i - N1 - N2; int o = j / 192, c = j % 192;
        w1t[j] = (bf16)w1[c * 768 + o];
    } else if (i < N1 + N2 + N3 + N3) {
        int j = i - N1 - N2 - N3; int o = j / 768, c = j % 768;
        w2t[j] = (bf16)w2[c * 192 + o];
    }
}

// ---------------- fused window attention (one WG per window) ----------------
__global__ __launch_bounds__(256) void attn_kernel(const float* __restrict__ x,
                                                   const bf16*  __restrict__ qkv_wt,
                                                   const float* __restrict__ qkv_b,
                                                   const bf16*  __restrict__ proj_wt,
                                                   const float* __restrict__ proj_b,
                                                   const float* __restrict__ g1,
                                                   const float* __restrict__ be1,
                                                   float* __restrict__ y) {
    __shared__ bf16  s_xn[NPAD * DIM] __attribute__((aligned(16)));   // xn, reused as attn output
    __shared__ bf16  s_q [NH * NPAD * HD] __attribute__((aligned(16)));
    __shared__ bf16  s_k [NH * NPAD * HD] __attribute__((aligned(16)));
    __shared__ bf16  s_vT[NH * HD * NPAD] __attribute__((aligned(16))); // [head][ch][token]
    __shared__ float s_S [NPAD * NPAD];
    __shared__ bf16  s_P [NPAD * NPAD] __attribute__((aligned(16)));

    const int lane = threadIdx.x & 31;
    const int wid  = threadIdx.x >> 5;
    const int widx = blockIdx.x;
    const int bimg = widx >> 6;
    const int wrem = widx & 63;
    const int wh   = wrem >> 3;
    const int wwi  = wrem & 7;

    // ---- gather (shift folded in) + LayerNorm -> bf16 LDS ----
    for (int r = wid; r < NPAD; r += 8) {
        if (r < NTOK) {
            int tr = r / WS_, tc = r % WS_;
            int sh = (wh * WS_ + tr + SS_) % HH;
            int sw = (wwi * WS_ + tc + SS_) % WW_;
            const float* src = x + ((size_t)(bimg * 3136 + sh * 56 + sw)) * DIM;
            float v[6], s = 0.f, sq = 0.f;
#pragma unroll
            for (int j = 0; j < 6; ++j) { float t = src[lane + j * 32]; v[j] = t; s += t; sq += t * t; }
#pragma unroll
            for (int off = 16; off; off >>= 1) { s += __shfl_xor(s, off, 32); sq += __shfl_xor(sq, off, 32); }
            float mu = s * (1.f / DIM);
            float rstd = rsqrtf(sq * (1.f / DIM) - mu * mu + 1e-5f);
#pragma unroll
            for (int j = 0; j < 6; ++j) {
                int c = lane + j * 32;
                s_xn[r * DIM + c] = (bf16)((v[j] - mu) * rstd * g1[c] + be1[c]);
            }
        } else {
#pragma unroll
            for (int j = 0; j < 6; ++j) s_xn[r * DIM + lane + j * 32] = (bf16)0.f;
        }
    }
    __syncthreads();

    // ---- QKV GEMM: [64 x 192] @ [192 x 576] ----
    const int n15 = lane & 15;
    for (int t = wid; t < 4 * 36; t += 8) {
        int row0 = (t / 36) * 16, col0 = (t % 36) * 16;
        v8f acc = {};
#pragma unroll
        for (int kc = 0; kc < 6; ++kc) {
            v16bf a = load_a(s_xn, DIM, row0, kc * 32, lane);
            v16bf b = load_b(qkv_wt, DIM, col0, kc * 32, lane);
            acc = wmma_bf16(a, b, acc);
        }
        // which-section / head are uniform per tile (16-wide tile never crosses a
        // q/k/v boundary (192) or a head boundary (32)) -> scalar branch, no exec dance.
        int sec = col0 / DIM;            // 0=q, 1=k, 2=v   (wave-uniform)
        int o0  = col0 % DIM;            // uniform
        int h   = o0 >> 5;               // uniform
        int ch  = (o0 & 31) + n15;       // per-lane
        float bi = qkv_b[col0 + n15];
        int rb   = row0 + (lane >> 4) * 8;
        if (sec == 2) {
            // transposed v: 8 consecutive tokens -> one 16B LDS store
            bf16 buf[8];
#pragma unroll
            for (int r2 = 0; r2 < 8; ++r2) buf[r2] = (bf16)(acc[r2] + bi);
            *reinterpret_cast<uint4*>(&s_vT[(h * HD + ch) * NPAD + rb]) =
                *reinterpret_cast<const uint4*>(buf);
        } else {
            bf16* dst = (sec == 0) ? s_q : s_k;
#pragma unroll
            for (int r2 = 0; r2 < 8; ++r2)
                dst[(h * NPAD + rb + r2) * HD + ch] = (bf16)(acc[r2] + bi);
        }
    }
    __syncthreads();

    const float scale = 0.17677669529663687f;   // 1/sqrt(32)
    for (int h = 0; h < NH; ++h) {
        // S = q @ k^T  (K = 32, single WMMA per 16x16 tile)
        for (int t = wid; t < 16; t += 8) {
            int row0 = (t >> 2) * 16, col0 = (t & 3) * 16;
            v16bf a = load_a(s_q + h * NPAD * HD, HD, row0, 0, lane);
            v16bf b = load_b(s_k + h * NPAD * HD, HD, col0, 0, lane);
            v8f acc = {};
            acc = wmma_bf16(a, b, acc);
            int n  = col0 + n15;
            int rb = row0 + (lane >> 4) * 8;
            bool oob = (n >= NTOK);
#pragma unroll
            for (int r2 = 0; r2 < 8; ++r2)
                s_S[(rb + r2) * NPAD + n] = oob ? -1e30f : acc[r2] * scale;
        }
        __syncthreads();
        // masked softmax, rows split 8 per wave
#pragma unroll
        for (int j = 0; j < 8; ++j) {
            int r = wid * 8 + j;
            float v0 = s_S[r * NPAD + lane];
            float v1 = s_S[r * NPAD + lane + 32];
            float m = fmaxf(v0, v1);
#pragma unroll
            for (int off = 16; off; off >>= 1) m = fmaxf(m, __shfl_xor(m, off, 32));
            float e0 = __expf(v0 - m), e1 = __expf(v1 - m);
            float s = e0 + e1;
#pragma unroll
            for (int off = 16; off; off >>= 1) s += __shfl_xor(s, off, 32);
            float inv = 1.f / s;
            s_P[r * NPAD + lane]      = (bf16)(e0 * inv);
            s_P[r * NPAD + lane + 32] = (bf16)(e1 * inv);
        }
        __syncthreads();
        // O = P @ v   (K = 64 tokens: 2 chunks; 8 tiles -> 1 per wave)
        {
            int row0 = (wid >> 1) * 16, col0 = (wid & 1) * 16;
            v8f acc = {};
#pragma unroll
            for (int kc = 0; kc < 2; ++kc) {
                v16bf a = load_a(s_P, NPAD, row0, kc * 32, lane);
                v16bf b = load_b(s_vT + h * HD * NPAD, NPAD, col0, kc * 32, lane);
                acc = wmma_bf16(a, b, acc);
            }
            int n  = col0 + n15;
            int rb = row0 + (lane >> 4) * 8;
#pragma unroll
            for (int r2 = 0; r2 < 8; ++r2)
                s_xn[(rb + r2) * DIM + h * HD + n] = (bf16)acc[r2];
        }
        __syncthreads();
    }

    // ---- proj GEMM + scatter (reverse shift) to y ----
    for (int t = wid; t < 4 * 12; t += 8) {
        int row0 = (t / 12) * 16, col0 = (t % 12) * 16;
        v8f acc = {};
#pragma unroll
        for (int kc = 0; kc < 6; ++kc) {
            v16bf a = load_a(s_xn, DIM, row0, kc * 32, lane);
            v16bf b = load_b(proj_wt, DIM, col0, kc * 32, lane);
            acc = wmma_bf16(a, b, acc);
        }
        int n    = col0 + n15;
        float bi = proj_b[n];
        int rb   = row0 + (lane >> 4) * 8;
#pragma unroll
        for (int r2 = 0; r2 < 8; ++r2) {
            int tok = rb + r2;
            if (tok < NTOK) {
                int tr = tok / WS_, tc = tok % WS_;
                int dh = (wh * WS_ + tr + SS_) % HH;
                int dw = (wwi * WS_ + tc + SS_) % WW_;
                y[((size_t)(bimg * 3136 + dh * 56 + dw)) * DIM + n] = acc[r2] + bi;
            }
        }
    }
}

// ---------------- fused LN + MLP + residual (16 tokens per wave) ----------------
__global__ __launch_bounds__(128) void mlp_kernel(const float* __restrict__ y,
                                                  const bf16*  __restrict__ w1t,
                                                  const float* __restrict__ b1,
                                                  const bf16*  __restrict__ w2t,
                                                  const float* __restrict__ b2,
                                                  const float* __restrict__ g2,
                                                  const float* __restrict__ be2,
                                                  float* __restrict__ out) {
    __shared__ bf16 s_xn[4][16 * DIM] __attribute__((aligned(16)));
    __shared__ bf16 s_h [4][16 * 32]  __attribute__((aligned(16)));
    __shared__ bf16 s_w1[32 * DIM]    __attribute__((aligned(16)));   // fc1 chunk: [32 out][192 in]
    __shared__ bf16 s_w2[DIM * 32]    __attribute__((aligned(16)));   // fc2 chunk: [192 out][32 in]

    const int lane = threadIdx.x & 31;
    const int wid  = threadIdx.x >> 5;
    const size_t tok0 = (size_t)blockIdx.x * 64 + wid * 16;
    bf16* xn  = s_xn[wid];
    bf16* hst = s_h[wid];

    // LayerNorm: lane pair (l, l+16) handles row l, 96 channels each
    {
        int row = lane & 15, half = lane >> 4;
        const float* src = y + (tok0 + row) * DIM + half * 96;
        float s = 0.f, sq = 0.f;
        for (int j = 0; j < 96; ++j) { float t = src[j]; s += t; sq += t * t; }
        s  += __shfl_xor(s, 16, 32);
        sq += __shfl_xor(sq, 16, 32);
        float mu = s * (1.f / DIM);
        float rstd = rsqrtf(sq * (1.f / DIM) - mu * mu + 1e-5f);
        for (int j = 0; j < 96; ++j) {
            int c = half * 96 + j;
            xn[row * DIM + c] = (bf16)((src[j] - mu) * rstd * g2[c] + be2[c]);
        }
    }

    // hoist xn A-fragments (6 K-chunks)
    v16bf afrag[6];
#pragma unroll
    for (int kc = 0; kc < 6; ++kc) afrag[kc] = load_a(xn, DIM, 0, kc * 32, lane);

    // persistent output accumulators, init with b2
    const int n15 = lane & 15;
    v8f acc[12];
#pragma unroll
    for (int ct = 0; ct < 12; ++ct) {
        float bb = b2[ct * 16 + n15];
        v8f t;
#pragma unroll
        for (int r2 = 0; r2 < 8; ++r2) t[r2] = bb;
        acc[ct] = t;
    }

    const int rb = (lane >> 4) * 8;
    for (int hc = 0; hc < 24; ++hc) {
        // ---- stage this hidden chunk's weights in LDS (async global->LDS) ----
        {
            const bf16* gw1 = w1t + (size_t)hc * 32 * DIM;     // contiguous 12 KB
#pragma unroll
            for (int j = threadIdx.x; j < 768; j += 128)
                copy16_g2l(gw1 + j * 8, s_w1 + j * 8);
            const bf16* gw2 = w2t + hc * 32;                   // 192 rows x 64 B
#pragma unroll
            for (int j = threadIdx.x; j < 768; j += 128) {
                int n = j >> 2, q4 = j & 3;
                copy16_g2l(gw2 + (size_t)n * HIDDEN + q4 * 8, s_w2 + n * 32 + q4 * 8);
            }
            stage_fence();
        }
        // ---- fc1 chunk (16x32) + exact GeLU -> per-wave LDS stage ----
#pragma unroll
        for (int ct2 = 0; ct2 < 2; ++ct2) {
            v8f a1 = {};
#pragma unroll
            for (int kc = 0; kc < 6; ++kc) {
                v16bf b = load_b(s_w1, DIM, ct2 * 16, kc * 32, lane);
                a1 = wmma_bf16(afrag[kc], b, a1);
            }
            float bi = b1[hc * 32 + ct2 * 16 + n15];
#pragma unroll
            for (int r2 = 0; r2 < 8; ++r2) {
                float t = a1[r2] + bi;
                t = 0.5f * t * (1.f + erff(t * 0.70710678118654752f));
                hst[(rb + r2) * 32 + ct2 * 16 + n15] = (bf16)t;
            }
        }
        // ---- fc2: accumulate this 32-wide hidden chunk into all 12 output tiles ----
        v16bf ah = load_a(hst, 32, 0, 0, lane);
#pragma unroll
        for (int ct = 0; ct < 12; ++ct) {
            v16bf b = load_b(s_w2, 32, ct * 16, 0, lane);
            acc[ct] = wmma_bf16(ah, b, acc[ct]);
        }
        __syncthreads();   // all waves done reading staged weights before next overwrite
    }

    // residual + store
#pragma unroll
    for (int ct = 0; ct < 12; ++ct) {
        int n = ct * 16 + n15;
#pragma unroll
        for (int r2 = 0; r2 < 8; ++r2) {
            size_t tok = tok0 + rb + r2;
            out[tok * DIM + n] = acc[ct][r2] + y[tok * DIM + n];
        }
    }
}

// ---------------- host launcher ----------------
extern "C" void kernel_launch(void* const* d_in, const int* in_sizes, int n_in,
                              void* d_out, int out_size, void* d_ws, size_t ws_size,
                              hipStream_t stream) {
    const float* x      = (const float*)d_in[0];
    const float* qkv_w  = (const float*)d_in[1];
    const float* qkv_b  = (const float*)d_in[2];
    const float* proj_w = (const float*)d_in[3];
    const float* proj_b = (const float*)d_in[4];
    const float* g1     = (const float*)d_in[5];
    const float* be1    = (const float*)d_in[6];
    const float* g2     = (const float*)d_in[7];
    const float* be2    = (const float*)d_in[8];
    const float* w1     = (const float*)d_in[9];
    const float* b1     = (const float*)d_in[10];
    const float* w2     = (const float*)d_in[11];
    const float* b2     = (const float*)d_in[12];

    char* ws = (char*)d_ws;
    float* y = (float*)ws;
    size_t yb = (size_t)BIMG * 3136 * DIM * sizeof(float);   // 77,070,336 B (16B aligned)
    bf16* qkv_wt  = (bf16*)(ws + yb);
    bf16* proj_wt = qkv_wt + 576 * 192;
    bf16* w1t     = proj_wt + 192 * 192;
    bf16* w2t     = w1t + 768 * 192;

    prep_weights_kernel<<<(442368 + 255) / 256, 256, 0, stream>>>(
        qkv_w, proj_w, w1, w2, qkv_wt, proj_wt, w1t, w2t);
    attn_kernel<<<NWIN, 256, 0, stream>>>(
        x, qkv_wt, qkv_b, proj_wt, proj_b, g1, be1, y);
    mlp_kernel<<<(BIMG * 3136) / 64, 128, 0, stream>>>(
        y, w1t, b1, w2t, b2, g2, be2, (float*)d_out);
}
